// SelfAttentionModule_62620623175753
// MI455X (gfx1250) — matrix operand order repaired and checked by reading, not compiled, not run on hardware
//
#include <hip/hip_runtime.h>

// ---------------------------------------------------------------------------
// Self-attention (B=4, S=4096, D=512) for gfx1250 / MI455X.
// bf16 WMMA (v_wmma_f32_16x16x32_bf16) for all three matmuls, fp32 softmax,
// flash-attention single pass (no 256 MiB score tensor), online softmax,
// global_prefetch_b8 ahead-of-use prefetch of the next key block.
// ---------------------------------------------------------------------------

typedef __bf16 v16bf __attribute__((ext_vector_type(16)));
typedef float  v8f   __attribute__((ext_vector_type(8)));

#define NB     4
#define SEQ    4096
#define DF     512
#define SCALE  0.044194173824159216f   // 1/sqrt(512)

union FragBF { v16bf v; unsigned int u[8]; };

__device__ __forceinline__ unsigned short f2bf(float f) {
    unsigned int u = __float_as_uint(f);
    unsigned int r = 0x7FFFu + ((u >> 16) & 1u);   // round-to-nearest-even
    return (unsigned short)((u + r) >> 16);
}

// A-fragment: 16x32 bf16 tile, row-major, lane holds row m=lane&15.
// ISA layout: VGPR v<4 -> K = 2v+lo+8*half ; v>=4 -> K = 16+2(v-4)+lo+8*half
__device__ __forceinline__ v16bf load_frag_a(const unsigned short* tile, int ld) {
    int lane = threadIdx.x & 31;
    int m = lane & 15, h = (lane >> 4) & 1;
    const unsigned int* row = (const unsigned int*)(tile + m * ld);
    FragBF f;
    int o = h * 4;
    f.u[0] = row[o + 0]; f.u[1] = row[o + 1];
    f.u[2] = row[o + 2]; f.u[3] = row[o + 3];
    f.u[4] = row[o + 8]; f.u[5] = row[o + 9];
    f.u[6] = row[o + 10]; f.u[7] = row[o + 11];
    return f.v;
}

// B-fragment: want B[k][n] = X[n][k] with X a 16x32 row-major tile
// (lane holds row n=lane&15).  ISA layout: v<4 -> K = 2v+lo+8*half ; v>=4 -> K = 16+2(v-4)+lo+8*half
__device__ __forceinline__ v16bf load_frag_b(const unsigned short* tile, int ld) {
    int lane = threadIdx.x & 31;
    int n = lane & 15, h = (lane >> 4) & 1;
    const unsigned int* row = (const unsigned int*)(tile + n * ld);
    FragBF f;
    int o = h * 4;
#pragma unroll
    for (int v = 0; v < 8; ++v) f.u[v] = row[o + v + (v >= 4 ? 4 : 0)];
    return f.v;
}

__device__ __forceinline__ v8f wmma_bf16(v16bf a, v16bf b, v8f c) {
    return __builtin_amdgcn_wmma_f32_16x16x32_bf16(
        /*neg_a=*/false, a, /*neg_b=*/false, b,
        /*c_mod=*/(short)0, c, /*reuse_a=*/false, /*reuse_b=*/false);
}

// ---------------------------------------------------------------------------
// K0: theta -> bf16 row-major, bf16 transpose (per batch), and the fp32 copy
//     into out[..., 512:1024] (the concat second half).
// ---------------------------------------------------------------------------
__global__ void __launch_bounds__(256)
prep_theta(const float* __restrict__ theta, float* __restrict__ out,
           unsigned short* __restrict__ tb, unsigned short* __restrict__ vt) {
    long long i = (long long)blockIdx.x * blockDim.x + threadIdx.x;
    const long long N = (long long)NB * SEQ * DF;
    if (i >= N) return;
    int d = (int)(i % DF);
    long long bs = i / DF;
    int s = (int)(bs % SEQ);
    int b = (int)(bs / SEQ);
    float x = theta[i];
    unsigned short hv = f2bf(x);
    tb[i] = hv;
    vt[((long long)b * DF + d) * SEQ + s] = hv;                 // theta^T (bf16)
    out[((long long)b * SEQ + s) * (2 * DF) + DF + d] = x;      // concat tail
}

__global__ void __launch_bounds__(256)
prep_w(const float* __restrict__ w, unsigned short* __restrict__ wb) {
    int i = blockIdx.x * blockDim.x + threadIdx.x;
    if (i < DF * DF) wb[i] = f2bf(w[i]);
}

// ---------------------------------------------------------------------------
// K1: Q = theta @ W^T + b  (bf16 WMMA, fp32 accum, bf16 store).
//     Each wave owns one 16x64 output tile; 1024 blocks x 8 waves.
// ---------------------------------------------------------------------------
__global__ void __launch_bounds__(256)
proj_q(const unsigned short* __restrict__ tb, const unsigned short* __restrict__ wb,
       const float* __restrict__ bq, unsigned short* __restrict__ qb) {
    int wave = threadIdx.x >> 5;
    int g = blockIdx.x * 8 + wave;        // 8192 wave-tiles
    int mt = g >> 3;                      // 0..1023  (M = 16384 rows)
    int ng = g & 7;                       // 0..7     (N = 512 in 64-col groups)
    int m0 = mt * 16, n0 = ng * 64;
    int lane = threadIdx.x & 31;
    int h = (lane >> 4) & 1, ln = lane & 15;

    const v8f vz = {};
    v8f acc[4];
#pragma unroll
    for (int t = 0; t < 4; ++t) acc[t] = vz;

    for (int k = 0; k < DF; k += 32) {
        v16bf a = load_frag_a(tb + (long long)m0 * DF + k, DF);
#pragma unroll
        for (int t = 0; t < 4; ++t) {
            v16bf bb = load_frag_b(wb + (long long)(n0 + t * 16) * DF + k, DF);
            acc[t] = wmma_bf16(a, bb, acc[t]);
        }
    }
#pragma unroll
    for (int t = 0; t < 4; ++t) {
        float bias = bq[n0 + t * 16 + ln];
#pragma unroll
        for (int r = 0; r < 8; ++r) {
            int m = m0 + r + 8 * h;
            qb[(long long)m * DF + n0 + t * 16 + ln] = f2bf(acc[t][r] + bias);
        }
    }
}

// ---------------------------------------------------------------------------
// K2: flash attention.  Block = 32 query rows of one batch, 8 waves arranged
//     as 2 query-strips x 4 groups.  Per 32-key step: each wave computes a
//     partial S over its 128-wide k-slice (8 WMMAs), partials summed via LDS,
//     identical fp32 online softmax per strip, P restaged via per-wave LDS to
//     A-frag layout, then 8 WMMAs accumulate the wave's 16x128 fp32 O slice.
//     Next key block's K/V lines are prefetched while this block computes.
// ---------------------------------------------------------------------------
__global__ void __launch_bounds__(256)
flash_attn(const unsigned short* __restrict__ qb,
           const unsigned short* __restrict__ vt,
           float* __restrict__ out) {
    __shared__ float Sp[2][4][2][8][32];          // 16 KiB partial-S exchange
    __shared__ unsigned short Pst[8][16][32];     // 8 KiB per-wave P staging

    int blk = blockIdx.x;                 // 0..511
    int b = blk >> 7;
    int q0 = (blk & 127) * 32;
    int wave = threadIdx.x >> 5;
    int strip = wave >> 2, grp = wave & 3;
    int lane = threadIdx.x & 31;
    int h = (lane >> 4) & 1, ln = lane & 15;

    const unsigned short* Qb = qb + (long long)b * SEQ * DF;
    const unsigned short* Vb = vt + (long long)b * DF * SEQ;   // theta^T
    int qrow0 = q0 + strip * 16;

    // Preload this wave's Q A-fragments (its 128-wide k-slice): 4 frags.
    v16bf qa[4];
#pragma unroll
    for (int t = 0; t < 4; ++t)
        qa[t] = load_frag_a(Qb + (long long)qrow0 * DF + grp * 128 + t * 32, DF);

    const v8f vz = {};
    v8f o[8];                              // 16 rows x 128 dims fp32 accum
#pragma unroll
    for (int i = 0; i < 8; ++i) o[i] = vz;
    float mrow[8], lrow[8];
#pragma unroll
    for (int r = 0; r < 8; ++r) { mrow[r] = -__builtin_inff(); lrow[r] = 0.f; }

    for (int j = 0; j < SEQ / 32; ++j) {
        int key0 = j * 32;

        // Partial S = Q[:,ks] * K[:,ks]^T  (keys are Q rows too)
        v8f s0 = vz, s1 = vz;
#pragma unroll
        for (int t = 0; t < 4; ++t) {
            int k = grp * 128 + t * 32;
            v16bf b0 = load_frag_b(Qb + (long long)key0 * DF + k, DF);
            v16bf b1 = load_frag_b(Qb + (long long)(key0 + 16) * DF + k, DF);
            s0 = wmma_bf16(qa[t], b0, s0);
            s1 = wmma_bf16(qa[t], b1, s1);
        }

        // Prefetch next key block while softmax/PV of this block proceeds.
        // Uniform branch: EXEC stays all-ones for the WMMAs below.
        int keyn = key0 + 32;
        if (keyn < SEQ) {
            // K rows: 32 rows, this wave's 128-elem (256 B) k-slice -> 2 lines.
            const unsigned short* pk =
                Qb + (long long)(keyn & ~31) * 0 +     // (keep types simple)
                (long long)( (keyn) + lane ) * DF + grp * 128;
            __builtin_prefetch(pk, 0, 1);
            __builtin_prefetch(pk + 64, 0, 1);
            // V rows: this wave's 128 dims x 64 B slice -> 4 rows per lane.
#pragma unroll
            for (int r4 = 0; r4 < 4; ++r4) {
                const unsigned short* pv =
                    Vb + (long long)(grp * 128 + r4 * 32 + lane) * SEQ + keyn;
                __builtin_prefetch(pv, 0, 1);
            }
        }

        __syncthreads();                   // WAR on Sp from previous iter
#pragma unroll
        for (int r = 0; r < 8; ++r) {
            Sp[strip][grp][0][r][lane] = s0[r];
            Sp[strip][grp][1][r][lane] = s1[r];
        }
        __syncthreads();

        float f0[8], f1[8];
#pragma unroll
        for (int r = 0; r < 8; ++r) {
            float a0 = 0.f, a1 = 0.f;
#pragma unroll
            for (int g = 0; g < 4; ++g) {
                a0 += Sp[strip][g][0][r][lane];
                a1 += Sp[strip][g][1][r][lane];
            }
            f0[r] = a0 * SCALE;
            f1[r] = a1 * SCALE;
        }

        // Online softmax (rows live in 16-lane halves; xor<=8 stays in-half)
        float alpha[8], p0[8], p1[8];
#pragma unroll
        for (int r = 0; r < 8; ++r) {
            float mx = fmaxf(f0[r], f1[r]);
            mx = fmaxf(mx, __shfl_xor(mx, 1));
            mx = fmaxf(mx, __shfl_xor(mx, 2));
            mx = fmaxf(mx, __shfl_xor(mx, 4));
            mx = fmaxf(mx, __shfl_xor(mx, 8));
            float mnew = fmaxf(mrow[r], mx);
            p0[r] = __expf(f0[r] - mnew);
            p1[r] = __expf(f1[r] - mnew);
            float rs = p0[r] + p1[r];
            rs += __shfl_xor(rs, 1);
            rs += __shfl_xor(rs, 2);
            rs += __shfl_xor(rs, 4);
            rs += __shfl_xor(rs, 8);
            alpha[r] = __expf(mrow[r] - mnew);
            lrow[r] = lrow[r] * alpha[r] + rs;
            mrow[r] = mnew;
        }
#pragma unroll
        for (int nt = 0; nt < 8; ++nt)
#pragma unroll
            for (int r = 0; r < 8; ++r) o[nt][r] *= alpha[r];

        // Stage P (C-layout -> bf16 A-layout) through this wave's LDS tile.
#pragma unroll
        for (int r = 0; r < 8; ++r) {
            Pst[wave][r + 8 * h][ln]      = f2bf(p0[r]);
            Pst[wave][r + 8 * h][16 + ln] = f2bf(p1[r]);
        }
        // Wave-local DS producer->consumer ordering (CDNA5 split counter).
        asm volatile("s_wait_dscnt 0" ::: "memory");
        v16bf pa = load_frag_a(&Pst[wave][0][0], 32);

        // O += P * V   (V rows come from theta^T -> plain row gather)
#pragma unroll
        for (int nt = 0; nt < 8; ++nt) {
            v16bf vb = load_frag_b(Vb + (long long)(grp * 128 + nt * 16) * SEQ + key0, SEQ);
            o[nt] = wmma_bf16(pa, vb, o[nt]);
        }
    }

    // Epilogue: divide by l, write out[..., 0:512] (row stride 1024 fp32).
#pragma unroll
    for (int nt = 0; nt < 8; ++nt) {
        int n = grp * 128 + nt * 16 + ln;
#pragma unroll
        for (int r = 0; r < 8; ++r) {
            int m = qrow0 + r + 8 * h;
            out[((long long)b * SEQ + m) * (2 * DF) + n] = o[nt][r] / lrow[r];
        }
    }
}

// ---------------------------------------------------------------------------
extern "C" void kernel_launch(void* const* d_in, const int* in_sizes, int n_in,
                              void* d_out, int out_size, void* d_ws, size_t ws_size,
                              hipStream_t stream) {
    const float* theta = (const float*)d_in[0];   // [4,4096,512] fp32
    const float* wq    = (const float*)d_in[1];   // [512,512] fp32
    const float* bq    = (const float*)d_in[2];   // [512] fp32
    float* out = (float*)d_out;                   // [4,4096,1024] fp32
    char* ws = (char*)d_ws;

    // Workspace layout (48.5 MiB total):
    unsigned short* qb = (unsigned short*)(ws);                              // Q bf16, 16 MiB
    unsigned short* tb = (unsigned short*)(ws + (size_t)16 * 1024 * 1024);   // theta bf16, 16 MiB
    unsigned short* vt = (unsigned short*)(ws + (size_t)32 * 1024 * 1024);   // theta^T bf16, 16 MiB
    unsigned short* wb = (unsigned short*)(ws + (size_t)48 * 1024 * 1024);   // W bf16, 512 KiB

    const long long N = (long long)NB * SEQ * DF;
    prep_theta<<<(int)((N + 255) / 256), 256, 0, stream>>>(theta, out, tb, vt);
    prep_w<<<(DF * DF + 255) / 256, 256, 0, stream>>>(wq, wb);
    proj_q<<<1024, 256, 0, stream>>>(tb, wb, bq, qb);
    flash_attn<<<512, 256, 0, stream>>>(qb, vt, out);
}
